// Encoder_7834020348402
// MI455X (gfx1250) — matrix operand order repaired and checked by reading, not compiled, not run on hardware
//
#include <hip/hip_runtime.h>

// Problem constants (match reference)
#define Bsz 128
#define Tn  64
#define Nf  128
#define Mh  256
#define FM  1024   // 4*M
#define M2  512    // 2*M

typedef __attribute__((ext_vector_type(16))) __bf16 v16bf;
typedef __attribute__((ext_vector_type(8)))  float  v8f;

union ABFrag { v16bf v; unsigned int u[8]; };

__device__ __forceinline__ unsigned short f2bf(float x) {
    unsigned int u = __float_as_uint(x);
    unsigned int r = u + 0x7FFFu + ((u >> 16) & 1u);   // round-to-nearest-even
    return (unsigned short)(r >> 16);
}

__device__ __forceinline__ float fast_sigmoid(float x) {
    x = fminf(fmaxf(x, -30.f), 30.f);
    return 1.0f / (1.0f + __expf(-x));
}
__device__ __forceinline__ float fast_tanh(float x) {
    x = fminf(fmaxf(x, -15.f), 15.f);
    float e = __expf(2.0f * x);
    return (e - 1.0f) / (e + 1.0f);
}

// ---- WMMA fragment loaders (layouts per CDNA5 ISA 7.12.2, wave32) ----------
// A 16x32 bf16 from row-major (K contiguous): lane holds row = lane&15;
// VGPR v holds K pair: k(v) = 16*(v>>2) + 8*(lane>>4) + 2*(v&3)
__device__ __forceinline__ v16bf load_a_rowmajor(const unsigned short* __restrict__ Sa,
                                                 int row, int lda, int kbase) {
    int lane = threadIdx.x & 31;
    int half = lane >> 4;
    const unsigned int* p = (const unsigned int*)(Sa + row * lda);
    ABFrag a;
#pragma unroll
    for (int v = 0; v < 8; ++v) {
        int k = kbase + 16 * (v >> 2) + 8 * half + 2 * (v & 3);
        a.u[v] = p[k >> 1];
    }
    return a.v;
}

// A with strided K (on-the-fly transpose): element A[row][k] = base[k*kstride + row]
__device__ __forceinline__ v16bf load_a_strided(const unsigned short* __restrict__ base,
                                                int row, int kstride, int kbase) {
    int lane = threadIdx.x & 31;
    int half = lane >> 4;
    ABFrag a;
#pragma unroll
    for (int v = 0; v < 8; ++v) {
        int k = kbase + 16 * (v >> 2) + 8 * half + 2 * (v & 3);
        unsigned int lo = base[(k)     * kstride + row];
        unsigned int hi = base[(k + 1) * kstride + row];
        a.u[v] = lo | (hi << 16);
    }
    return a.v;
}

// B fragment from PRE-PACKED layout: for tile (kt, ct) each lane owns 8
// contiguous dwords -> two aligned b128 loads.
__device__ __forceinline__ v16bf load_b_packed(const unsigned int* __restrict__ Pb,
                                               int nkt, int kt, int ct) {
    int lane = threadIdx.x & 31;
    const unsigned int* p = Pb + ((((ct * nkt) + kt) * 32 + lane) << 3);
    ABFrag b;
#pragma unroll
    for (int v = 0; v < 8; ++v) b.u[v] = p[v];
    return b.v;
}

#define WMMA_BF16(A, Bm, C) \
    __builtin_amdgcn_wmma_f32_16x16x32_bf16(false, (A), false, (Bm), (short)0, (C), false, false)

// ---- Setup kernels ---------------------------------------------------------
__global__ void k_cvt_bf16(const float* __restrict__ src, unsigned short* __restrict__ dst, int n) {
    int i = blockIdx.x * blockDim.x + threadIdx.x;
    if (i < n) dst[i] = f2bf(src[i]);
}

// Pack a (K x C) row-major f32 weight into WMMA B-fragment order, fused bf16 cvt.
// dst dword i: v=i&7, lane=(i>>3)&31, tile=i>>8; kt=tile%nkt, ct=tile/nkt;
//   k = kt*32 + 16*(v>>2) + 8*(lane>>4) + 2*(v&3) ; col = ct*16 + (lane&15)
__global__ void k_pack_b(const float* __restrict__ src, unsigned int* __restrict__ dst,
                         int K, int C) {
    int i = blockIdx.x * blockDim.x + threadIdx.x;
    int nkt = K >> 5;
    int total = nkt * (C >> 4) * 256;
    if (i >= total) return;
    int v    = i & 7;
    int lane = (i >> 3) & 31;
    int tile = i >> 8;
    int kt = tile % nkt;
    int ct = tile / nkt;
    int k   = kt * 32 + 16 * (v >> 2) + 8 * (lane >> 4) + 2 * (v & 3);
    int col = ct * 16 + (lane & 15);
    unsigned int lo = f2bf(src[(k)     * C + col]);
    unsigned int hi = f2bf(src[(k + 1) * C + col]);
    dst[i] = lo | (hi << 16);
}

__global__ void k_init_state(const float* __restrict__ h0, const float* __restrict__ c0,
                             float* __restrict__ cbuf, unsigned short* __restrict__ hc0) {
    int i = blockIdx.x * blockDim.x + threadIdx.x;
    if (i < Bsz * Mh) {
        int b = i / Mh, m = i % Mh;
        cbuf[i] = c0[i];
        hc0[b * M2 + m]      = f2bf(h0[i]);
        hc0[b * M2 + Mh + m] = f2bf(c0[i]);
    }
}

// ---- xW = data @ W + b  : (B*T, FM), K = N ---------------------------------
__global__ void k_xw(const unsigned short* __restrict__ Abf,  // (B*T, N) bf16
                     const unsigned int* __restrict__ Wp,     // packed (N x FM)
                     const float* __restrict__ bias,          // (FM)
                     float* __restrict__ xW) {                // (B*T, FM)
    int wid  = blockIdx.x * (blockDim.x >> 5) + (threadIdx.x >> 5);
    int lane = threadIdx.x & 31;
    int tile_r = wid >> 6;                 // 512 row tiles
    int tile_c = wid & 63;                 // 64 col tiles
    int row = tile_r * 16 + (lane & 15);
    int col = tile_c * 16 + (lane & 15);
    v8f acc = {0.f, 0.f, 0.f, 0.f, 0.f, 0.f, 0.f, 0.f};
#pragma unroll
    for (int kt = 0; kt < (Nf >> 5); ++kt) {
        v16bf a  = load_a_rowmajor(Abf, row, Nf, kt << 5);
        v16bf bm = load_b_packed(Wp, Nf >> 5, kt, tile_c);
        acc = WMMA_BF16(a, bm, acc);
    }
    int half = lane >> 4;
    float bc = bias[col];
#pragma unroll
    for (int v = 0; v < 8; ++v) {
        int r = tile_r * 16 + v + 8 * half;
        xW[r * FM + col] = acc[v] + bc;
    }
}

// ---- w1x[b] = data[b]^T @ w1_k + w1_b : (B, N, T), K = T -------------------
__global__ void k_w1x(const unsigned short* __restrict__ dbf,   // (B,T,N) bf16
                      const unsigned int* __restrict__ w1kp,    // packed (T x T)
                      const float* __restrict__ w1b,            // (T)
                      float* __restrict__ w1x) {                // (B,N,T)
    int wid  = blockIdx.x * (blockDim.x >> 5) + (threadIdx.x >> 5);
    int lane = threadIdx.x & 31;
    int b      = wid >> 5;       // 32 tiles per batch
    int tidx   = wid & 31;
    int tile_f = tidx >> 2;      // 8 feature tiles
    int tile_s = tidx & 3;       // 4 time tiles
    int frow = tile_f * 16 + (lane & 15);
    int col  = tile_s * 16 + (lane & 15);
    const unsigned short* base = dbf + b * Tn * Nf;  // A[f][t] = base[t*N + f]
    v8f acc = {0.f, 0.f, 0.f, 0.f, 0.f, 0.f, 0.f, 0.f};
#pragma unroll
    for (int kt = 0; kt < (Tn >> 5); ++kt) {
        v16bf a  = load_a_strided(base, frow, Nf, kt << 5);
        v16bf bm = load_b_packed(w1kp, Tn >> 5, kt, tile_s);
        acc = WMMA_BF16(a, bm, acc);
    }
    int half = lane >> 4;
    float bc = w1b[col];
#pragma unroll
    for (int v = 0; v < 8; ++v) {
        int f = tile_f * 16 + v + 8 * half;
        w1x[(b * Nf + f) * Tn + col] = acc[v] + bc;
    }
}

// ---- Per-step LSTM cell: z = xW[:,t,:] + h@U ; gates ; h2,c2 ---------------
// One wave per 16x16 tile of (B x M); 4 accumulators (i,f,g,o) share A-fragment.
__global__ void k_gates(const float* __restrict__ xW,            // (B*T, FM)
                        const unsigned short* __restrict__ hc_in,// (B, 2M) bf16 (h in cols 0..M-1)
                        const unsigned int* __restrict__ Up,     // packed (M x FM)
                        float* __restrict__ cbuf,                // (B, M) f32 cell state
                        unsigned short* __restrict__ hc_out,     // (B, 2M) bf16
                        int t) {
    int wid  = blockIdx.x * (blockDim.x >> 5) + (threadIdx.x >> 5);
    int lane = threadIdx.x & 31;
    int tile_b = wid >> 4;   // 8 batch tiles
    int tile_m = wid & 15;   // 16 m tiles
    int brow = tile_b * 16 + (lane & 15);
    int mcol = tile_m * 16 + (lane & 15);
    int half = lane >> 4;
    const int nkt = Mh >> 5;  // 8

    v8f ai, af, ag, ao;
#pragma unroll
    for (int v = 0; v < 8; ++v) {
        int bb = tile_b * 16 + v + 8 * half;
        const float* zr = xW + (bb * Tn + t) * FM;
        ai[v] = zr[0 * Mh + mcol];
        af[v] = zr[1 * Mh + mcol];
        ag[v] = zr[2 * Mh + mcol];
        ao[v] = zr[3 * Mh + mcol];
    }
#pragma unroll
    for (int kt = 0; kt < nkt; ++kt) {
        v16bf a  = load_a_rowmajor(hc_in, brow, M2, kt << 5);   // h part only (k < M)
        v16bf b0 = load_b_packed(Up, nkt, kt, 0 * 16 + tile_m);
        ai = WMMA_BF16(a, b0, ai);
        v16bf b1 = load_b_packed(Up, nkt, kt, 1 * 16 + tile_m);
        af = WMMA_BF16(a, b1, af);
        v16bf b2 = load_b_packed(Up, nkt, kt, 2 * 16 + tile_m);
        ag = WMMA_BF16(a, b2, ag);
        v16bf b3 = load_b_packed(Up, nkt, kt, 3 * 16 + tile_m);
        ao = WMMA_BF16(a, b3, ao);
    }
#pragma unroll
    for (int v = 0; v < 8; ++v) {
        int bb = tile_b * 16 + v + 8 * half;
        float gi = fast_sigmoid(ai[v]);
        float gf = fast_sigmoid(af[v]);
        float gg = fast_tanh(ag[v]);
        float go = fast_sigmoid(ao[v]);
        float cold = cbuf[bb * Mh + mcol];
        float c2 = gf * cold + gi * gg;
        float h2 = go * fast_tanh(c2);
        cbuf[bb * Mh + mcol] = c2;
        hc_out[bb * M2 + mcol]      = f2bf(h2);
        hc_out[bb * M2 + Mh + mcol] = f2bf(c2);
    }
}

// ---- q = [h2,c2] @ w2_k + w2_b : (B, T), K = 2M ----------------------------
__global__ void k_q(const unsigned short* __restrict__ hc,     // (B, 2M) bf16
                    const unsigned int* __restrict__ w2kp,     // packed (2M x T)
                    const float* __restrict__ w2b,             // (T)
                    float* __restrict__ q) {                   // (B, T)
    int wid  = blockIdx.x * (blockDim.x >> 5) + (threadIdx.x >> 5);
    int lane = threadIdx.x & 31;
    int tile_b = wid >> 2;   // 8
    int tile_s = wid & 3;    // 4
    int brow = tile_b * 16 + (lane & 15);
    int col  = tile_s * 16 + (lane & 15);
    v8f acc = {0.f, 0.f, 0.f, 0.f, 0.f, 0.f, 0.f, 0.f};
#pragma unroll
    for (int kt = 0; kt < (M2 >> 5); ++kt) {
        v16bf a  = load_a_rowmajor(hc, brow, M2, kt << 5);
        v16bf bm = load_b_packed(w2kp, M2 >> 5, kt, tile_s);
        acc = WMMA_BF16(a, bm, acc);
    }
    int half = lane >> 4;
    float bc = w2b[col];
#pragma unroll
    for (int v = 0; v < 8; ++v) {
        int bb = tile_b * 16 + v + 8 * half;
        q[bb * Tn + col] = acc[v] + bc;
    }
}

// ---- Attention: s = tanh(w1x + q)·v ; softmax over features ; out = data*alpha
__global__ void k_attn(const float* __restrict__ w1x,  // (B,N,T)
                       const float* __restrict__ q,    // (B,T)
                       const float* __restrict__ vk,   // (T)
                       const float* __restrict__ vb,   // (1)
                       const float* __restrict__ data, // (B,T,N)
                       float* __restrict__ out,        // (B,T,N)
                       int t) {
    __shared__ float wtile[Nf * Tn];     // 32 KB: w1x tile for this batch
    __shared__ float qs[Tn], vks[Tn], red[Nf];
    int b = blockIdx.x;
    int f = threadIdx.x;                  // 128 threads = 4 waves
    // coalesced LDS stage of w1x[b] (8192 floats) via float4
    {
        const float4* s4 = (const float4*)(w1x + b * Nf * Tn);
        float4* d4 = (float4*)wtile;
#pragma unroll
        for (int i = f; i < (Nf * Tn) / 4; i += Nf) d4[i] = s4[i];
    }
    if (f < Tn) { qs[f] = q[b * Tn + f]; vks[f] = vk[f]; }
    __syncthreads();
    const float* wr = wtile + f * Tn;
    float acc = vb[0];
#pragma unroll 8
    for (int s = 0; s < Tn; ++s)
        acc += fast_tanh(wr[s] + qs[s]) * vks[s];
    // softmax over the 128 features (one per thread)
    red[f] = acc; __syncthreads();
#pragma unroll
    for (int off = 64; off > 0; off >>= 1) {
        if (f < off) red[f] = fmaxf(red[f], red[f + off]);
        __syncthreads();
    }
    float mx = red[0]; __syncthreads();
    float e = __expf(acc - mx);
    red[f] = e; __syncthreads();
#pragma unroll
    for (int off = 64; off > 0; off >>= 1) {
        if (f < off) red[f] += red[f + off];
        __syncthreads();
    }
    float inv = 1.0f / red[0];
    int idx = (b * Tn + t) * Nf + f;
    out[idx] = data[idx] * (e * inv);
}

// ---- Host orchestration ----------------------------------------------------
extern "C" void kernel_launch(void* const* d_in, const int* in_sizes, int n_in,
                              void* d_out, int out_size, void* d_ws, size_t ws_size,
                              hipStream_t stream) {
    (void)in_sizes; (void)n_in; (void)out_size; (void)ws_size;
    const float* data = (const float*)d_in[0];
    const float* h0   = (const float*)d_in[1];
    const float* c0   = (const float*)d_in[2];
    const float* W    = (const float*)d_in[3];
    const float* U    = (const float*)d_in[4];
    const float* bb   = (const float*)d_in[5];
    const float* w1k  = (const float*)d_in[6];
    const float* w1b  = (const float*)d_in[7];
    const float* w2k  = (const float*)d_in[8];
    const float* w2b  = (const float*)d_in[9];
    const float* vk   = (const float*)d_in[10];
    const float* vb   = (const float*)d_in[11];
    float* out = (float*)d_out;

    // Workspace layout (bytes)
    char* ws = (char*)d_ws;
    size_t o = 0;
    float* xW   = (float*)(ws + o); o += (size_t)Bsz * Tn * FM * 4;   // 33.55 MB
    float* w1x  = (float*)(ws + o); o += (size_t)Bsz * Nf * Tn * 4;   //  4.19 MB
    float* cbuf = (float*)(ws + o); o += (size_t)Bsz * Mh * 4;
    float* qbuf = (float*)(ws + o); o += (size_t)Bsz * Tn * 4;
    unsigned short* hc0 = (unsigned short*)(ws + o); o += (size_t)Bsz * M2 * 2;
    unsigned short* hc1 = (unsigned short*)(ws + o); o += (size_t)Bsz * M2 * 2;
    unsigned short* dbf = (unsigned short*)(ws + o); o += (size_t)Bsz * Tn * Nf * 2;
    unsigned int* Wp    = (unsigned int*)(ws + o); o += (size_t)Nf * FM * 2;  // packed bf16 pairs
    unsigned int* Up    = (unsigned int*)(ws + o); o += (size_t)Mh * FM * 2;
    unsigned int* w1kp  = (unsigned int*)(ws + o); o += (size_t)Tn * Tn * 2;
    unsigned int* w2kp  = (unsigned int*)(ws + o); o += (size_t)M2 * Tn * 2;

    // Operand prep: activations -> bf16; weights -> packed WMMA-B layout (fused cvt)
    k_cvt_bf16<<<(Bsz * Tn * Nf + 255) / 256, 256, 0, stream>>>(data, dbf, Bsz * Tn * Nf);
    k_pack_b<<<(Nf * FM / 2 + 255) / 256, 256, 0, stream>>>(W, Wp, Nf, FM);
    k_pack_b<<<(Mh * FM / 2 + 255) / 256, 256, 0, stream>>>(U, Up, Mh, FM);
    k_pack_b<<<(Tn * Tn / 2 + 255) / 256, 256, 0, stream>>>(w1k, w1kp, Tn, Tn);
    k_pack_b<<<(M2 * Tn / 2 + 255) / 256, 256, 0, stream>>>(w2k, w2kp, M2, Tn);
    k_init_state<<<(Bsz * Mh + 255) / 256, 256, 0, stream>>>(h0, c0, cbuf, hc0);

    // Hoisted parallel GEMMs (all WMMA)
    k_xw<<<(512 * 64) / 4, 128, 0, stream>>>(dbf, Wp, bb, xW);       // 32768 waves
    k_w1x<<<(Bsz * 32) / 4, 128, 0, stream>>>(dbf, w1kp, w1b, w1x);  // 4096 waves

    // Sequential recurrence (double-buffered bf16 [h,c])
    for (int t = 0; t < Tn; ++t) {
        const unsigned short* hin = (t & 1) ? hc1 : hc0;
        unsigned short* hout      = (t & 1) ? hc0 : hc1;
        k_gates<<<(8 * 16) / 4, 128, 0, stream>>>(xW, hin, Up, cbuf, hout, t);
        k_q<<<(8 * 4) / 4, 128, 0, stream>>>(hout, w2kp, w2b, qbuf);
        k_attn<<<Bsz, Nf, 0, stream>>>(w1x, qbuf, vk, vb, data, out, t);
    }
}
